// MoEDispatcher_48584670052582
// MI455X (gfx1250) — compile-verified, matrix operand order checked by reading, not run on hardware
//
#include <hip/hip_runtime.h>
#include <cmath>

// ---------------------------------------------------------------------------
// MoE dispatcher for MI455X (gfx1250, wave32).
// Routed formulation: out[tok] = sum_k w_k * FFN_{e_k}(x[tok])
//   FFN_e(x) = silu(x @ w1[e] + b1[e]) @ w2[e] + b2[e]
// ~275 GFLOP routed compute vs ~0.5 GB HBM traffic -> WMMA-bound.
// fp16 operands + f32 WMMA accumulation (v_wmma_f32_16x16x32_f16).
// ffn2 A-stage uses GLOBAL_LOAD_ASYNC_TO_LDS_B128 (+ s_wait_asynccnt).
// ---------------------------------------------------------------------------

typedef __attribute__((ext_vector_type(16))) _Float16 v16h;
typedef __attribute__((ext_vector_type(8)))  _Float16 v8h;
typedef __attribute__((ext_vector_type(4)))  _Float16 v4h;
typedef __attribute__((ext_vector_type(8)))  float    v8f;

namespace {
constexpr int Bb = 4, Ll = 2048, Dd = 1024, Hh = 4096, Ee = 8, Kk = 2;
constexpr int NTOK    = Bb * Ll;      // 8192 tokens
constexpr int NASSIGN = NTOK * Kk;    // 16384 (token, expert) assignments
constexpr int CAP     = NASSIGN;      // worst-case tokens per expert
constexpr int TM = 128;               // token-tile per block
constexpr int TN = 128;               // output-col tile per block
constexpr int KC = 32;                // K-depth per LDS stage (= WMMA K)

// workspace layout (bytes)
constexpr size_t WS_COUNTS  = 0;                                          // E ints
constexpr size_t WS_CURSORS = 64;                                         // E ints
constexpr size_t WS_OFFSETS = 128;                                        // E ints
constexpr size_t WS_TOKID   = 256;                                        // NASSIGN ints
constexpr size_t WS_TOKW    = WS_TOKID + (size_t)NASSIGN * sizeof(int);   // NASSIGN floats
constexpr size_t WS_H       = (WS_TOKW + (size_t)NASSIGN * sizeof(float) + 255) & ~(size_t)255;
// h stage: NASSIGN * Hh fp16 = 128 MB
}

union HV { v16h v; v8h h[2]; };

__device__ __forceinline__ void wait_asynccnt0() {
#if __has_builtin(__builtin_amdgcn_s_wait_asynccnt)
  __builtin_amdgcn_s_wait_asynccnt(0);
#else
  asm volatile("s_wait_asynccnt 0x0" ::: "memory");
#endif
}

// ---------------------------------------------------------------------------
// Kernel 0: zero output accumulator + dispatch counters
// ---------------------------------------------------------------------------
__global__ void moe_init(float* __restrict__ out, int* __restrict__ counts,
                         int* __restrict__ cursors) {
  size_t i = ((size_t)blockIdx.x * blockDim.x + threadIdx.x) * 4;
  float4 z = {0.f, 0.f, 0.f, 0.f};
  *(float4*)(out + i) = z;                 // grid sized exactly: NTOK*Dd / 4 threads
  if (blockIdx.x == 0 && threadIdx.x < Ee) {
    counts[threadIdx.x]  = 0;
    cursors[threadIdx.x] = 0;
  }
}

// ---------------------------------------------------------------------------
// Kernel 1: histogram of assignments per expert
// ---------------------------------------------------------------------------
__global__ void moe_count(const int* __restrict__ idx, int* __restrict__ counts) {
  int i = blockIdx.x * blockDim.x + threadIdx.x;
  if (i < NASSIGN) atomicAdd(&counts[idx[i]], 1);
}

// ---------------------------------------------------------------------------
// Kernel 2: exclusive scan over 8 experts (single lane)
// ---------------------------------------------------------------------------
__global__ void moe_scan(const int* __restrict__ counts, int* __restrict__ offsets) {
  if (threadIdx.x == 0) {
    int s = 0;
    for (int e = 0; e < Ee; ++e) { offsets[e] = s; s += counts[e]; }
  }
}

// ---------------------------------------------------------------------------
// Kernel 3: scatter (tokenId, weight) into compact per-expert segments
// ---------------------------------------------------------------------------
__global__ void moe_scatter(const int* __restrict__ idx, const float* __restrict__ ew,
                            const int* __restrict__ offsets, int* __restrict__ cursors,
                            int* __restrict__ tokId, float* __restrict__ tokW) {
  int i = blockIdx.x * blockDim.x + threadIdx.x;
  if (i < NASSIGN) {
    int e   = idx[i];
    int pos = offsets[e] + atomicAdd(&cursors[e], 1);
    tokId[pos] = i / Kk;
    tokW[pos]  = ew[i];
  }
}

// ---------------------------------------------------------------------------
// Shared helper: stage a KC x TN fp32 weight tile into N-major fp16 LDS.
// thread t: 4 consecutive k rows x 4 consecutive n cols; packs 4 k-halfs
// per n into one ds_store_b64 (4 stores/thread instead of 16 b16 stores).
// ---------------------------------------------------------------------------
__device__ __forceinline__ void stage_w_tile(_Float16 (*Bs)[KC], const float* __restrict__ wrow,
                                             int ldw, int t) {
  const int kg = (t & 7) * 4;
  const int nb = (t >> 3) * 4;
  const float* s0 = wrow + (size_t)(kg + 0) * ldw + nb;
  const float* s1 = wrow + (size_t)(kg + 1) * ldw + nb;
  const float* s2 = wrow + (size_t)(kg + 2) * ldw + nb;
  const float* s3 = wrow + (size_t)(kg + 3) * ldw + nb;
  float4 r0 = *(const float4*)s0;
  float4 r1 = *(const float4*)s1;
  float4 r2 = *(const float4*)s2;
  float4 r3 = *(const float4*)s3;
  __builtin_prefetch(s0 + (size_t)KC * ldw, 0, 1);     // next K tile of weights
  v4h c0 = {(_Float16)r0.x, (_Float16)r1.x, (_Float16)r2.x, (_Float16)r3.x};
  v4h c1 = {(_Float16)r0.y, (_Float16)r1.y, (_Float16)r2.y, (_Float16)r3.y};
  v4h c2 = {(_Float16)r0.z, (_Float16)r1.z, (_Float16)r2.z, (_Float16)r3.z};
  v4h c3 = {(_Float16)r0.w, (_Float16)r1.w, (_Float16)r2.w, (_Float16)r3.w};
  *(v4h*)&Bs[nb + 0][kg] = c0;
  *(v4h*)&Bs[nb + 1][kg] = c1;
  *(v4h*)&Bs[nb + 2][kg] = c2;
  *(v4h*)&Bs[nb + 3][kg] = c3;
}

// ---------------------------------------------------------------------------
// Kernel 4: h = silu(gather(x) @ w1[e] + b1[e]) -> fp16 stage
// grid: (Hh/TN, CAP/TM, Ee); block: 256 threads = 8 waves (2x4 wave grid)
// ---------------------------------------------------------------------------
__global__ __launch_bounds__(256) void moe_ffn1(
    const float* __restrict__ x, const float* __restrict__ w1,
    const float* __restrict__ b1, const int* __restrict__ counts,
    const int* __restrict__ offsets, const int* __restrict__ tokId,
    _Float16* __restrict__ h) {
  const int e   = blockIdx.z;
  const int cnt = counts[e];
  const int rowBase = blockIdx.y * TM;
  if (rowBase >= cnt) return;                       // data-dependent early exit
  const int colBase = blockIdx.x * TN;              // over H
  const int offs    = offsets[e];

  __shared__ _Float16 As[TM][KC];                   // [row][k]
  __shared__ _Float16 Bs[TN][KC];                   // [n][k]  (w1 tile, N-major)
  __shared__ int rowTok[TM];

  const int t = threadIdx.x;
  if (t < TM) {
    int lr = rowBase + t;
    rowTok[t] = tokId[offs + ((lr < cnt) ? lr : 0)];
  }

  const int wave = t >> 5, lane = t & 31;
  const int wm = wave & 1, wn = wave >> 1;          // 2x4 wave grid, 64x32 each
  const int lhalf = lane >> 4, lmod = lane & 15;
  const int kb = lhalf * 8;                         // fragment K base per lane half

  v8f acc[4][2];
#pragma unroll
  for (int mf = 0; mf < 4; ++mf)
#pragma unroll
    for (int nf = 0; nf < 2; ++nf) acc[mf][nf] = (v8f)(0.f);

  const float* w1e = w1 + (size_t)e * Dd * Hh;

  for (int k0 = 0; k0 < Dd; k0 += KC) {
    __syncthreads();                                // protect As/Bs reuse + rowTok
    {   // stage A: 128 rows x 32 k of gathered x, cast fp32->fp16
      int row = t >> 1, kh = (t & 1) * 16;
      const float* src = x + (size_t)rowTok[row] * Dd + k0 + kh;
      float4 f0 = *(const float4*)(src);
      float4 f1 = *(const float4*)(src + 4);
      float4 f2 = *(const float4*)(src + 8);
      float4 f3 = *(const float4*)(src + 12);
      __builtin_prefetch(src + KC, 0, 1);           // next K tile
      v8h p0, p1;
      p0[0]=(_Float16)f0.x; p0[1]=(_Float16)f0.y; p0[2]=(_Float16)f0.z; p0[3]=(_Float16)f0.w;
      p0[4]=(_Float16)f1.x; p0[5]=(_Float16)f1.y; p0[6]=(_Float16)f1.z; p0[7]=(_Float16)f1.w;
      p1[0]=(_Float16)f2.x; p1[1]=(_Float16)f2.y; p1[2]=(_Float16)f2.z; p1[3]=(_Float16)f2.w;
      p1[4]=(_Float16)f3.x; p1[5]=(_Float16)f3.y; p1[6]=(_Float16)f3.z; p1[7]=(_Float16)f3.w;
      *(v8h*)&As[row][kh]     = p0;
      *(v8h*)&As[row][kh + 8] = p1;
    }
    // stage B: w1 tile [k0..k0+KC) x [colBase..+TN), N-major packed b64 stores
    stage_w_tile(Bs, w1e + (size_t)k0 * Hh + colBase, Hh, t);
    __syncthreads();

    v16h afr[4], bfr[2];
#pragma unroll
    for (int mf = 0; mf < 4; ++mf) {                // two ds_load_b128 per frag
      int row = wm * 64 + mf * 16 + lmod;
      HV u;
      u.h[0] = *(const v8h*)&As[row][kb];
      u.h[1] = *(const v8h*)&As[row][16 + kb];
      afr[mf] = u.v;
    }
#pragma unroll
    for (int nf = 0; nf < 2; ++nf) {
      int col = wn * 32 + nf * 16 + lmod;
      HV u;
      u.h[0] = *(const v8h*)&Bs[col][kb];
      u.h[1] = *(const v8h*)&Bs[col][16 + kb];
      bfr[nf] = u.v;
    }
#pragma unroll
    for (int mf = 0; mf < 4; ++mf)
#pragma unroll
      for (int nf = 0; nf < 2; ++nf)
        acc[mf][nf] = __builtin_amdgcn_wmma_f32_16x16x32_f16(
            false, afr[mf], false, bfr[nf], (short)0, acc[mf][nf], false, false);
  }

  // epilogue: bias + SiLU (fast v_rcp_f32 sigmoid) -> fp16 h stage
  const float* b1e = b1 + (size_t)e * Hh;
#pragma unroll
  for (int mf = 0; mf < 4; ++mf) {
#pragma unroll
    for (int nf = 0; nf < 2; ++nf) {
      int col = colBase + wn * 32 + nf * 16 + lmod;
      float bias = b1e[col];
#pragma unroll
      for (int r = 0; r < 8; ++r) {
        int m  = r + 8 * lhalf;                     // C layout: vgpr r, lane-half
        int lr = rowBase + wm * 64 + mf * 16 + m;
        if (lr < cnt) {
          float v = acc[mf][nf][r] + bias;
          v = v * __builtin_amdgcn_rcpf(1.0f + __expf(-v));   // silu, no div chain
          h[(size_t)(offs + lr) * Hh + col] = (_Float16)v;
        }
      }
    }
  }
}

// ---------------------------------------------------------------------------
// Kernel 5: out[tok] += w * (h @ w2[e] + b2[e])
// grid: (Dd/TN, CAP/TM, Ee)
// A-stage: async DMA global->LDS (global_load_async_to_lds_b128)
// ---------------------------------------------------------------------------
__global__ __launch_bounds__(256) void moe_ffn2(
    const _Float16* __restrict__ h, const float* __restrict__ w2,
    const float* __restrict__ b2, const int* __restrict__ counts,
    const int* __restrict__ offsets, const int* __restrict__ tokId,
    const float* __restrict__ tokW, float* __restrict__ out) {
  const int e   = blockIdx.z;
  const int cnt = counts[e];
  const int rowBase = blockIdx.y * TM;
  if (rowBase >= cnt) return;
  const int colBase = blockIdx.x * TN;              // over D
  const int offs    = offsets[e];

  __shared__ _Float16 As[TM][KC];
  __shared__ _Float16 Bs[TN][KC];

  const int t = threadIdx.x;
  const int wave = t >> 5, lane = t & 31;
  const int wm = wave & 1, wn = wave >> 1;
  const int lhalf = lane >> 4, lmod = lane & 15;
  const int kb = lhalf * 8;

  v8f acc[4][2];
#pragma unroll
  for (int mf = 0; mf < 4; ++mf)
#pragma unroll
    for (int nf = 0; nf < 2; ++nf) acc[mf][nf] = (v8f)(0.f);

  const float* w2e = w2 + (size_t)e * Hh * Dd;

  // per-thread A-stage addressing (fixed across K loop)
  const int arow = t >> 1, akh = (t & 1) * 16;
  {
    int lr = rowBase + arow;
    int entry = offs + ((lr < cnt) ? lr : (cnt - 1));
    const _Float16* asrc0 = h + (size_t)entry * Hh + akh;
    unsigned lds0 = (unsigned)(uintptr_t)&As[arow][akh];   // flat->LDS offset (low 32b)

    for (int k0 = 0; k0 < Hh; k0 += KC) {
      __syncthreads();                // prior iter's frag ds_loads are complete
      {   // stage A: 16 halfs/thread via async DMA to LDS (no VGPR roundtrip)
        unsigned long long ga = (unsigned long long)(uintptr_t)(asrc0 + k0);
        asm volatile(
            "global_load_async_to_lds_b128 %0, %1, off\n\t"
            "global_load_async_to_lds_b128 %0, %1, off offset:16"
            :: "v"(lds0), "v"(ga) : "memory");
      }
      // stage B: w2 tile, N-major packed b64 stores
      stage_w_tile(Bs, w2e + (size_t)k0 * Dd + colBase, Dd, t);
      wait_asynccnt0();               // async LDS writes visible before barrier
      __syncthreads();

      v16h afr[4], bfr[2];
#pragma unroll
      for (int mf = 0; mf < 4; ++mf) {
        int row = wm * 64 + mf * 16 + lmod;
        HV u;
        u.h[0] = *(const v8h*)&As[row][kb];
        u.h[1] = *(const v8h*)&As[row][16 + kb];
        afr[mf] = u.v;
      }
#pragma unroll
      for (int nf = 0; nf < 2; ++nf) {
        int col = wn * 32 + nf * 16 + lmod;
        HV u;
        u.h[0] = *(const v8h*)&Bs[col][kb];
        u.h[1] = *(const v8h*)&Bs[col][16 + kb];
        bfr[nf] = u.v;
      }
#pragma unroll
      for (int mf = 0; mf < 4; ++mf)
#pragma unroll
        for (int nf = 0; nf < 2; ++nf)
          acc[mf][nf] = __builtin_amdgcn_wmma_f32_16x16x32_f16(
              false, afr[mf], false, bfr[nf], (short)0, acc[mf][nf], false, false);
    }
  }

  // epilogue: bias, routing weight, accumulate (fp32 hardware atomics)
  const float* b2e = b2 + (size_t)e * Dd;
#pragma unroll
  for (int mf = 0; mf < 4; ++mf) {
#pragma unroll
    for (int r = 0; r < 8; ++r) {
      int m  = r + 8 * lhalf;
      int lr = rowBase + wm * 64 + mf * 16 + m;
      if (lr >= cnt) continue;
      int entry = offs + lr;
      int tok   = tokId[entry];
      float wgt = tokW[entry];
      float* orow = out + (size_t)tok * Dd;
#pragma unroll
      for (int nf = 0; nf < 2; ++nf) {
        int col = colBase + wn * 32 + nf * 16 + lmod;
        float v = (acc[mf][nf][r] + b2e[col]) * wgt;
        unsafeAtomicAdd(&orow[col], v);             // global_atomic_add_f32
      }
    }
  }
}

// ---------------------------------------------------------------------------
extern "C" void kernel_launch(void* const* d_in, const int* in_sizes, int n_in,
                              void* d_out, int out_size, void* d_ws, size_t ws_size,
                              hipStream_t stream) {
  (void)in_sizes; (void)n_in; (void)out_size; (void)ws_size;
  const float* x   = (const float*)d_in[0];
  const int*   idx = (const int*)d_in[1];   // expert_indices
  const float* ew  = (const float*)d_in[2]; // expert_weights
  const float* w1  = (const float*)d_in[3];
  const float* b1  = (const float*)d_in[4];
  const float* w2  = (const float*)d_in[5];
  const float* b2  = (const float*)d_in[6];
  float* out = (float*)d_out;

  char* ws = (char*)d_ws;
  int*   counts  = (int*)(ws + WS_COUNTS);
  int*   cursors = (int*)(ws + WS_CURSORS);
  int*   offsets = (int*)(ws + WS_OFFSETS);
  int*   tokId   = (int*)(ws + WS_TOKID);
  float* tokW    = (float*)(ws + WS_TOKW);
  _Float16* hbuf = (_Float16*)(ws + WS_H);

  moe_init<<<(NTOK * Dd) / (256 * 4), 256, 0, stream>>>(out, counts, cursors);
  moe_count<<<NASSIGN / 256, 256, 0, stream>>>(idx, counts);
  moe_scan<<<1, 32, 0, stream>>>(counts, offsets);
  moe_scatter<<<NASSIGN / 256, 256, 0, stream>>>(idx, ew, offsets, cursors, tokId, tokW);

  moe_ffn1<<<dim3(Hh / TN, CAP / TM, Ee), 256, 0, stream>>>(
      x, w1, b1, counts, offsets, tokId, hbuf);
  moe_ffn2<<<dim3(Dd / TN, CAP / TM, Ee), 256, 0, stream>>>(
      hbuf, w2, b2, counts, offsets, tokId, tokW, out);
}
